// BenchmarkFromHell_20572893348683
// MI455X (gfx1250) — compile-verified, matrix operand order checked by reading, not run on hardware
//
#include <hip/hip_runtime.h>
#include <hip/hip_bf16.h>
#include <math.h>

// ---------------------------------------------------------------------------
// Types for gfx1250 WMMA
// ---------------------------------------------------------------------------
typedef __bf16 bf16;
typedef __attribute__((ext_vector_type(16))) __bf16 bf16x16;
typedef __attribute__((ext_vector_type(8)))  __bf16 bf16x8;
typedef __attribute__((ext_vector_type(8)))  float  fx8;
typedef __attribute__((ext_vector_type(4)))  float  fx4;

// Native f32->bf16 conversion (gfx1250 has hw bf16 converts; clang lowers
// this to v_cvt_pk_bf16_f32 / v_cvt_bf16_f32 instead of a bit-twiddle chain).
__device__ __forceinline__ void cvt_store4(bf16* dst, fx4 v) {
    union { bf16 b[4]; unsigned long long u; } p;
    p.b[0] = (bf16)v[0]; p.b[1] = (bf16)v[1];
    p.b[2] = (bf16)v[2]; p.b[3] = (bf16)v[3];
    *(unsigned long long*)dst = p.u;               // 8B-aligned LDS store
}

// Load one 16x32 bf16 fragment row-slice per documented gfx1250 layout:
// lane holds k in {hi*8..hi*8+7} (elems 0..7) and {16+hi*8..23+hi*8} (elems 8..15)
__device__ __forceinline__ bf16x16 ldfrag(const bf16* rowbase_hi8) {
    union { bf16x16 v; bf16x8 h[2]; } u;
    u.h[0] = *(const bf16x8*)(rowbase_hi8);        // 16B aligned
    u.h[1] = *(const bf16x8*)(rowbase_hi8 + 16);
    return u.v;
}

// ---------------------------------------------------------------------------
// Kernel 1: _make_kernel  (one block, n = 200 or 3200)
// ---------------------------------------------------------------------------
__global__ __launch_bounds__(256) void makew_kernel(
    const float* __restrict__ base, const float* __restrict__ a,
    const float* __restrict__ b, const float* __restrict__ thrash,
    const float* __restrict__ noise, float* __restrict__ wout,
    int n, float S)
{
    __shared__ float kbuf[3200];
    __shared__ float ssum[128];
    __shared__ float red[256];
    __shared__ float tsum;
    const int t = threadIdx.x;
    const int slices = n / 25;

    for (int s = t; s < slices; s += 256) {
        float sum = 0.f;
        for (int j = 0; j < 25; j++) sum += base[s * 25 + j] * S;
        ssum[s] = sum;
    }
    // thrash gather + reduce (23 elements)
    float tv = 0.f;
    if (t < 23) {
        int TJ = (t * 7919) % 23;
        int TK = (TJ * 1543) % 23;
        tv = thrash[t * 529 + TJ * 23 + TK];
    }
    red[t] = tv; __syncthreads();
    for (int s = 128; s > 0; s >>= 1) { if (t < s) red[t] += red[t + s]; __syncthreads(); }
    if (t == 0) tsum = red[0] * 1e-12f;
    __syncthreads();

    const float ab = a[0] * b[0];
    for (int i = t; i < n; i += 256) {
        float acc = base[i] * S + ab * ssum[i / 25] + tsum;
        float r = noise[i];
        r = r * (r + 1e-7f); r = r * (r + 1e-7f); r = r * (r + 1e-7f);
        kbuf[i] = acc + r;
    }
    __syncthreads();
    float pa = 0.f;
    for (int i = t; i < n; i += 256) pa += fabsf(kbuf[i]);
    red[t] = pa; __syncthreads();
    for (int s = 128; s > 0; s >>= 1) { if (t < s) red[t] += red[t + s]; __syncthreads(); }
    const float denom = red[0] / (float)n + 1e-12f;
    for (int i = t; i < n; i += 256) wout[i] = kbuf[i] / denom;
}

// ---------------------------------------------------------------------------
// Kernel 2: conv1  x[256,1,28,28] * w1[8,1,5,5] pad2 -> relu -> /pool
// One block per sample.
// ---------------------------------------------------------------------------
__global__ __launch_bounds__(256) void conv1_kernel(
    const float* __restrict__ x, const float* __restrict__ w1,
    float* __restrict__ h1)
{
    __shared__ float xs[784];
    __shared__ float ws[200];
    __shared__ float outs[6272];
    __shared__ float red[256];
    const int b = blockIdx.x, t = threadIdx.x;
    for (int i = t; i < 784; i += 256) xs[i] = x[b * 784 + i];
    for (int i = t; i < 200; i += 256) ws[i] = w1[i];
    __syncthreads();
    for (int idx = t; idx < 6272; idx += 256) {
        int o = idx / 784, p = idx % 784, py = p / 28, px = p % 28;
        float acc = 0.f;
        for (int dy = 0; dy < 5; dy++) {
            int yy = py + dy - 2; if (yy < 0 || yy >= 28) continue;
            for (int dx = 0; dx < 5; dx++) {
                int xx = px + dx - 2; if (xx < 0 || xx >= 28) continue;
                acc += xs[yy * 28 + xx] * ws[o * 25 + dy * 5 + dx];
            }
        }
        outs[idx] = acc;
    }
    __syncthreads();
    float pa = 0.f;
    for (int i = t; i < 6272; i += 256) pa += outs[i];
    red[t] = pa; __syncthreads();
    for (int s = 128; s > 0; s >>= 1) { if (t < s) red[t] += red[t + s]; __syncthreads(); }
    const float mean = red[0] / 6272.0f;
    const float invpool = 1.0f / (784.0f + 1e-9f);
    for (int i = t; i < 6272; i += 256) {
        float v = outs[i] + mean * 1e-20f;
        v = v > 0.f ? v : 0.f;
        h1[b * 6272 + i] = v * invpool;
    }
}

// ---------------------------------------------------------------------------
// Kernel 3: conv2 + relu + pool + row-normalize + v*(v+1e-12)
// One block per sample.
// ---------------------------------------------------------------------------
__global__ __launch_bounds__(256) void conv2_kernel(
    const float* __restrict__ h1, const float* __restrict__ w2,
    float* __restrict__ V)
{
    __shared__ float hs[6272];
    __shared__ float ws[3200];
    __shared__ float outs[12544];
    __shared__ float red[256];
    const int b = blockIdx.x, t = threadIdx.x;
    for (int i = t; i < 6272; i += 256) hs[i] = h1[b * 6272 + i];
    for (int i = t; i < 3200; i += 256) ws[i] = w2[i];
    __syncthreads();
    for (int idx = t; idx < 12544; idx += 256) {
        int o = idx / 784, p = idx % 784, py = p / 28, px = p % 28;
        float acc = 0.f;
        for (int ic = 0; ic < 8; ic++) {
            const float* hp = &hs[ic * 784];
            const float* wp = &ws[o * 200 + ic * 25];
            for (int dy = 0; dy < 5; dy++) {
                int yy = py + dy - 2; if (yy < 0 || yy >= 28) continue;
                for (int dx = 0; dx < 5; dx++) {
                    int xx = px + dx - 2; if (xx < 0 || xx >= 28) continue;
                    acc += hp[yy * 28 + xx] * wp[dy * 5 + dx];
                }
            }
        }
        outs[idx] = acc;
    }
    __syncthreads();
    float pa = 0.f;
    for (int i = t; i < 12544; i += 256) pa += outs[i];
    red[t] = pa; __syncthreads();
    for (int s = 128; s > 0; s >>= 1) { if (t < s) red[t] += red[t + s]; __syncthreads(); }
    const float mean = red[0] / 12544.0f;
    const float invpool = 1.0f / (784.0f + 1e-9f);
    __syncthreads();
    for (int i = t; i < 12544; i += 256) {
        float v = outs[i] + mean * 1e-20f;
        v = v > 0.f ? v : 0.f;
        outs[i] = v * invpool;
    }
    __syncthreads();
    float ss = 0.f;
    for (int i = t; i < 12544; i += 256) ss += outs[i] * outs[i];
    red[t] = ss; __syncthreads();
    for (int s = 128; s > 0; s >>= 1) { if (t < s) red[t] += red[t + s]; __syncthreads(); }
    const float inv = 1.0f / (sqrtf(red[0]) + 1e-20f);
    for (int i = t; i < 12544; i += 256) {
        float vn = outs[i] * inv;
        V[(size_t)b * 12544 + i] = vn * (vn + 1e-12f);
    }
}

// ---------------------------------------------------------------------------
// Kernel 4: the 80.5 GFLOP GEMM  U[256,12544] = V[256,12544] @ W[12544,12544]^T
// bf16 WMMA 16x16x32, BLOCK 256x64x32, double-buffered LDS, 8 waves,
// each wave: 32 rows x 64 cols = 2x4 tiles -> 8 v_wmma per K-step.
// W is read from HBM exactly once (tiles partition N x K).
// ---------------------------------------------------------------------------
#define GEMM_K 12544
#define BM 256
#define BN 64
#define BK 32
#define LDT 40   // 32 + 8 bf16 pad: bank-stride 20 dwords -> conflict-free b128

__global__ __launch_bounds__(256) void gemm_wmma(
    const float* __restrict__ V, const float* __restrict__ W,
    float* __restrict__ U)
{
    __shared__ bf16 As[2][BM][LDT];   // m x k
    __shared__ bf16 Bs[2][BN][LDT];   // n x k
    const int tid = threadIdx.x;
    const int lane = tid & 31, wave = tid >> 5;
    const int hi = lane >> 4, l15 = lane & 15;
    const int n0 = blockIdx.x * BN;

    fx8 acc[2][4];
#pragma unroll
    for (int i = 0; i < 2; i++)
#pragma unroll
        for (int j = 0; j < 4; j++) {
            fx8 z = {0.f,0.f,0.f,0.f,0.f,0.f,0.f,0.f};
            acc[i][j] = z;
        }

    fx4 ra[8], rb[2];
    // ---- prologue: stage k-tile 0 ----
#pragma unroll
    for (int c = 0; c < 8; c++) {
        int g = tid + 256 * c, row = g >> 3, ch = g & 7;
        ra[c] = *(const fx4*)(V + (size_t)row * GEMM_K + ch * 4);
    }
#pragma unroll
    for (int c = 0; c < 2; c++) {
        int g = tid + 256 * c, row = g >> 3, ch = g & 7;
        rb[c] = *(const fx4*)(W + (size_t)(n0 + row) * GEMM_K + ch * 4);
    }
#pragma unroll
    for (int c = 0; c < 8; c++) {
        int g = tid + 256 * c, row = g >> 3, ch = g & 7;
        cvt_store4(&As[0][row][ch * 4], ra[c]);
    }
#pragma unroll
    for (int c = 0; c < 2; c++) {
        int g = tid + 256 * c, row = g >> 3, ch = g & 7;
        cvt_store4(&Bs[0][row][ch * 4], rb[c]);
    }
    __syncthreads();

    const int NT = GEMM_K / BK;   // 392
    for (int t = 0; t < NT; t++) {
        const int cur = t & 1;
        // prefetch next K-tile into registers (overlaps with WMMA below)
        if (t + 1 < NT) {
            const int k0n = (t + 1) * BK;
#pragma unroll
            for (int c = 0; c < 8; c++) {
                int g = tid + 256 * c, row = g >> 3, ch = g & 7;
                ra[c] = *(const fx4*)(V + (size_t)row * GEMM_K + k0n + ch * 4);
            }
#pragma unroll
            for (int c = 0; c < 2; c++) {
                int g = tid + 256 * c, row = g >> 3, ch = g & 7;
                rb[c] = *(const fx4*)(W + (size_t)(n0 + row) * GEMM_K + k0n + ch * 4);
            }
        }
        // fragment loads + 8 WMMAs
        bf16x16 af[2], bfr[4];
#pragma unroll
        for (int mf = 0; mf < 2; mf++)
            af[mf] = ldfrag(&As[cur][wave * 32 + mf * 16 + l15][hi * 8]);
#pragma unroll
        for (int nf = 0; nf < 4; nf++)
            bfr[nf] = ldfrag(&Bs[cur][nf * 16 + l15][hi * 8]);
#pragma unroll
        for (int mf = 0; mf < 2; mf++)
#pragma unroll
            for (int nf = 0; nf < 4; nf++)
                acc[mf][nf] = __builtin_amdgcn_wmma_f32_16x16x32_bf16(
                    false, af[mf], false, bfr[nf], (short)0, acc[mf][nf],
                    false, false);
        __syncthreads();
        if (t + 1 < NT) {
            const int nxt = 1 - cur;
#pragma unroll
            for (int c = 0; c < 8; c++) {
                int g = tid + 256 * c, row = g >> 3, ch = g & 7;
                cvt_store4(&As[nxt][row][ch * 4], ra[c]);
            }
#pragma unroll
            for (int c = 0; c < 2; c++) {
                int g = tid + 256 * c, row = g >> 3, ch = g & 7;
                cvt_store4(&Bs[nxt][row][ch * 4], rb[c]);
            }
        }
        __syncthreads();
    }

    // epilogue: C/D layout -> lanes 0-15: M=r, lanes 16-31: M=r+8
#pragma unroll
    for (int mf = 0; mf < 2; mf++)
#pragma unroll
        for (int nf = 0; nf < 4; nf++)
#pragma unroll
            for (int r = 0; r < 8; r++) {
                int m = wave * 32 + mf * 16 + hi * 8 + r;
                int n = n0 + nf * 16 + l15;
                U[(size_t)m * GEMM_K + n] = acc[mf][nf][r];
            }
}

// ---------------------------------------------------------------------------
// Kernel 5: fc  y[256,10] = U @ fc_w.T + fc_b  (+ global abs-sum via atomic)
// ---------------------------------------------------------------------------
__global__ __launch_bounds__(256) void fc_kernel(
    const float* __restrict__ U, const float* __restrict__ fw,
    const float* __restrict__ fb, float* __restrict__ yraw,
    float* __restrict__ sumabs)
{
    __shared__ float red[256];
    const int b = blockIdx.x, t = threadIdx.x;
    float p[10];
#pragma unroll
    for (int j = 0; j < 10; j++) p[j] = 0.f;
    for (int k = t; k < 12544; k += 256) {
        float uv = U[(size_t)b * 12544 + k];
#pragma unroll
        for (int j = 0; j < 10; j++) p[j] += uv * fw[j * 12544 + k];
    }
    float blockabs = 0.f;
    for (int j = 0; j < 10; j++) {
        red[t] = p[j]; __syncthreads();
        for (int s = 128; s > 0; s >>= 1) { if (t < s) red[t] += red[t + s]; __syncthreads(); }
        if (t == 0) {
            float val = red[0] + fb[j];
            yraw[b * 10 + j] = val;
            blockabs += fabsf(val);
        }
        __syncthreads();
    }
    if (t == 0) atomicAdd(sumabs, blockabs);
}

__global__ void zero1_kernel(float* p) { p[0] = 0.f; }

__global__ void scale_kernel(const float* __restrict__ yraw,
                             const float* __restrict__ sum,
                             float* __restrict__ out)
{
    int i = blockIdx.x * blockDim.x + threadIdx.x;
    if (i < 2560) {
        float mean = sum[0] / 2560.0f;
        out[i] = yraw[i] / (mean + 1e-30f);
    }
}

// ---------------------------------------------------------------------------
// Host launcher
// ---------------------------------------------------------------------------
extern "C" void kernel_launch(void* const* d_in, const int* in_sizes, int n_in,
                              void* d_out, int out_size, void* d_ws, size_t ws_size,
                              hipStream_t stream) {
    (void)in_sizes; (void)n_in; (void)out_size; (void)ws_size;
    const float* x       = (const float*)d_in[0];
    const float* base1   = (const float*)d_in[1];
    const float* a1      = (const float*)d_in[2];
    const float* b1      = (const float*)d_in[3];
    const float* thrash1 = (const float*)d_in[4];
    const float* noise1  = (const float*)d_in[5];
    const float* base2   = (const float*)d_in[6];
    const float* a2      = (const float*)d_in[7];
    const float* b2      = (const float*)d_in[8];
    const float* thrash2 = (const float*)d_in[9];
    const float* noise2  = (const float*)d_in[10];
    const float* lin_w   = (const float*)d_in[11];
    const float* fc_w    = (const float*)d_in[12];
    const float* fc_b    = (const float*)d_in[13];
    float* out = (float*)d_out;
    float* ws  = (float*)d_ws;

    // scratch layout (floats)
    const size_t W1_OFF   = 0;                          // 200
    const size_t W2_OFF   = 256;                        // 3200
    const size_t H1_OFF   = 4096;                       // 256*8*784  = 1605632
    const size_t V_OFF    = H1_OFF + 1605632;           // 256*12544  = 3211264
    const size_t U_OFF    = V_OFF + 3211264;            // 256*12544  = 3211264
    const size_t YRAW_OFF = U_OFF + 3211264;            // 2560
    const size_t SUM_OFF  = YRAW_OFF + 2560;            // 1  (total ~32.1 MB)

    // _S computed on host in double precision (matches python float64 math)
    double SC = 0, SD = 0, SE = 0;
    for (int c = 0; c < 5; c++) SC += sin((double)(c + 1));
    for (int d = 0; d < 5; d++) SD += 1.0 / (cos((double)d + 1e-9) + 1e-9);
    for (int e = 0; e < 5; e++) SE += sqrt((double)(e + 1));
    const float S = (float)(SC * SD * SE);

    makew_kernel<<<1, 256, 0, stream>>>(base1, a1, b1, thrash1, noise1,
                                        ws + W1_OFF, 200, S);
    makew_kernel<<<1, 256, 0, stream>>>(base2, a2, b2, thrash2, noise2,
                                        ws + W2_OFF, 3200, S);
    conv1_kernel<<<256, 256, 0, stream>>>(x, ws + W1_OFF, ws + H1_OFF);
    conv2_kernel<<<256, 256, 0, stream>>>(ws + H1_OFF, ws + W2_OFF, ws + V_OFF);
    gemm_wmma<<<12544 / BN, 256, 0, stream>>>(ws + V_OFF, lin_w, ws + U_OFF);
    zero1_kernel<<<1, 1, 0, stream>>>(ws + SUM_OFF);
    fc_kernel<<<256, 256, 0, stream>>>(ws + U_OFF, fc_w, fc_b,
                                       ws + YRAW_OFF, ws + SUM_OFF);
    scale_kernel<<<10, 256, 0, stream>>>(ws + YRAW_OFF, ws + SUM_OFF, out);
}
